// RingAttention_42649025249824
// MI455X (gfx1250) — matrix operand order repaired and checked by reading, not compiled, not run on hardware
//
#include <hip/hip_runtime.h>

#define DEVFN __device__ __forceinline__

typedef unsigned short u16;
typedef unsigned int   u32;
typedef unsigned long long u64;
typedef __attribute__((ext_vector_type(16))) __bf16 v16bf;
typedef __attribute__((ext_vector_type(8)))  float  v8f;
typedef __attribute__((ext_vector_type(4)))  u32    u32x4;
typedef __attribute__((ext_vector_type(8)))  int    i32x8;
typedef __attribute__((ext_vector_type(4)))  int    i32x4;

// ---- CDNA5 feature gates (device pass only; host pass parses fallbacks) ----
#if defined(__HIP_DEVICE_COMPILE__) && \
    __has_builtin(__builtin_amdgcn_tensor_load_to_lds) && \
    __has_builtin(__builtin_amdgcn_s_wait_tensorcnt)
#define HAVE_TDM 1
#if __clang_major__ >= 23
#define TDM_6ARG 1            // therock/clang-23: 6-arg builtin
#else
#define TDM_6ARG 0            // ROCm 7.2/clang-22: 5-arg builtin
#endif
#else
#define HAVE_TDM 0
#endif

#if defined(__HIP_DEVICE_COMPILE__) && \
    __has_builtin(__builtin_amdgcn_global_load_async_to_lds_b128) && \
    __has_builtin(__builtin_amdgcn_s_wait_asynccnt)
#define HAVE_ASYNC 1
#else
#define HAVE_ASYNC 0
#endif

#if HAVE_ASYNC
typedef __attribute__((address_space(1))) i32x4 as1_i32x4;  // global v4i
typedef __attribute__((address_space(3))) i32x4 as3_i32x4;  // LDS v4i
#endif

union Frag {          // 8 VGPRs = 16 bf16 per lane (A or B operand of 16x16x32)
  u32x4 u[2];
  v16bf v;
};

static constexpr int kB  = 2;
static constexpr int kS  = 2048;
static constexpr int kH  = 1024;     // HID
static constexpr int kNH = 16;
static constexpr int kHD = 64;
static constexpr int kM  = kB * kS;  // 4096 rows of x

static constexpr float kBIG = 3.0e38f;

DEVFN u16 f2bf(float f) {            // round-to-nearest-even fp32 -> bf16
  u32 u = __float_as_uint(f);
  u += 0x7FFFu + ((u >> 16) & 1u);
  return (u16)(u >> 16);
}

DEVFN v8f wmma_bf16(const Frag& a, const Frag& b, v8f c) {
  return __builtin_amdgcn_wmma_f32_16x16x32_bf16(
      false, a.v, false, b.v, (short)0, c, false, false);
}

// 16B per-lane async global->LDS copy (ASYNCcnt path), plain copy fallback
DEVFN void async_cp16(const u16* g, u16* l) {
#if HAVE_ASYNC
  __builtin_amdgcn_global_load_async_to_lds_b128(
      (as1_i32x4*)(u64)g, (as3_i32x4*)(u32)(u64)l, 0, 0);
#else
  *(u32x4*)l = *(const u32x4*)g;
#endif
}

DEVFN void async_wait() {
#if HAVE_ASYNC
  __builtin_amdgcn_s_wait_asynccnt(0);
#endif
}

#if HAVE_TDM
// 2D bf16 tile DMA: global (row stride stride_w elems) -> LDS with TDM padding.
// pad_interval code: row bytes = 1<<code qwords; pad_amount code 3 = 4 DWORDs
// = 16B, matching LDS row stride = tile_w*2 + 16 bytes.
DEVFN void tdm_tile_load(const u16* g, const u16* l,
                         u32 tile_w, u32 tile_h,
                         u32 tensor_w, u32 tensor_h, u32 stride_w,
                         u32 pad_interval_code) {
  u64 ga = (u64)g;
  u32 la = (u32)(u64)l;                               // LDS byte offset
  u32x4 g0 = { 1u,                                    // count=1, user mode
               la,                                    // lds_addr [63:32]
               (u32)ga,                                // global_addr [95:64]
               (u32)(ga >> 32) | (2u << 30) };         // addr[56:32] | type=2
  i32x8 g1 = {
    (int)((1u << 16)                                  // data_size = 2B
          | (1u << 20)                                // pad_enable
          | (pad_interval_code << 22)                 // pad every row
          | (3u << 25)),                              // pad_amount = 4 DWORDs
    (int)((tensor_w & 0xFFFFu) << 16),                // tensor_dim0 lo16 [63:48]
    (int)((tensor_w >> 16) | ((tensor_h & 0xFFFFu) << 16)),  // d0 hi | d1 lo
    (int)((tensor_h >> 16) | (tile_w << 16)),         // d1 hi | tile_dim0
    (int)(tile_h & 0xFFFFu),                          // tile_dim1 (tile_dim2=0)
    (int)stride_w,                                    // tensor_dim0_stride lo32
    0, 0 };
  i32x4 gz = { 0, 0, 0, 0 };                          // 2D: groups 2/3 zero
#if TDM_6ARG
  i32x8 z8 = { 0, 0, 0, 0, 0, 0, 0, 0 };
  __builtin_amdgcn_tensor_load_to_lds(g0, g1, gz, gz, z8, 0);
#else
  __builtin_amdgcn_tensor_load_to_lds(g0, g1, gz, gz, 0);
#endif
}
#endif

// ---------------------------------------------------------------- fp32->bf16
__global__ __launch_bounds__(256) void cvt_bf16_kernel(
    const float* __restrict__ in, u16* __restrict__ out, int n4) {
  int i = blockIdx.x * blockDim.x + threadIdx.x;
  if (i < n4) {
    float4 f = ((const float4*)in)[i];
    uint2 o;
    o.x = (u32)f2bf(f.x) | ((u32)f2bf(f.y) << 16);
    o.y = (u32)f2bf(f.z) | ((u32)f2bf(f.w) << 16);
    ((uint2*)out)[i] = o;
  }
}

// ------------------------------------------------------- tiled WMMA bf16 GEMM
// C[m,n] = sum_k A[m,k] * W[n,k]   (W stored [out,in] so both K-contiguous)
// mode 0: fp32 linear [kM][kH] out.  mode 1: bf16 scatter to [B][NH][S][HD].
// Double-buffered LDS; tiles staged by TDM (or async-to-LDS fallback).
static constexpr int BM = 128, BN = 128, BK = 32;
static constexpr int LDT = BK + 8;            // 40 halfs = 80B rows (16B aligned)

__global__ __launch_bounds__(256) void gemm_bf16_kernel(
    const u16* __restrict__ A, const u16* __restrict__ W,
    float* __restrict__ outF, u16* __restrict__ outS, int mode) {
  __shared__ u16 As[2][BM * LDT];
  __shared__ u16 Bs[2][BN * LDT];

  const int tid  = threadIdx.x;
  const int wave = tid >> 5;
  const int lane = tid & 31;
  const int lm   = lane & 15;
  const int lh   = lane >> 4;
  const int wm0  = (wave >> 2) * 64;   // 2 wave-rows
  const int wn0  = (wave & 3) * 32;    // 4 wave-cols
  const int m0   = blockIdx.y * BM;
  const int n0   = blockIdx.x * BN;

  v8f acc[4][2] = {};

  auto issue = [&](int k0, int buf) {
#if HAVE_TDM
    if (wave == 0) {   // one wave drives the Tensor Data Mover (EXEC ignored)
      tdm_tile_load(&A[(size_t)m0 * kH + k0], &As[buf][0], BK, BM, kH, kM, kH, 3);
      tdm_tile_load(&W[(size_t)n0 * kH + k0], &Bs[buf][0], BK, BN, kH, kH, kH, 3);
    }
#else
    #pragma unroll
    for (int j = 0; j < 2; ++j) {      // 512 x 16B segments, 2 per thread
      int seg = tid + j * 256;
      int row = seg >> 2;
      int c8  = (seg & 3) * 8;
      async_cp16(&A[(m0 + row) * kH + k0 + c8], &As[buf][row * LDT + c8]);
      async_cp16(&W[(n0 + row) * kH + k0 + c8], &Bs[buf][row * LDT + c8]);
    }
#endif
  };
  auto wait_tiles = [&]() {
#if HAVE_TDM
    if (wave == 0) __builtin_amdgcn_s_wait_tensorcnt(0);
#else
    async_wait();
#endif
    __syncthreads();
  };

  issue(0, 0);
  wait_tiles();

  int cur = 0;
  for (int k0 = 0; k0 < kH; k0 += BK) {
    if (k0 + BK < kH) issue(k0 + BK, cur ^ 1);   // overlap DMA with WMMAs

    const u16* as = &As[cur][0];
    const u16* bs = &Bs[cur][0];
    Frag af[4], bfr[2];
    #pragma unroll
    for (int i = 0; i < 4; ++i) {     // A frag: m = lm, K split by lane-half
      const u16* p = &as[(wm0 + i * 16 + lm) * LDT];
      af[i].u[0] = *(const u32x4*)(p + 8 * lh);
      af[i].u[1] = *(const u32x4*)(p + 16 + 8 * lh);
    }
    #pragma unroll
    for (int j = 0; j < 2; ++j) {     // B frag: n = lm, K split by lane-half
      const u16* p = &bs[(wn0 + j * 16 + lm) * LDT + 8 * lh];
      bfr[j].u[0] = *(const u32x4*)(p);
      bfr[j].u[1] = *(const u32x4*)(p + 16);
    }
    #pragma unroll
    for (int i = 0; i < 4; ++i)
      #pragma unroll
      for (int j = 0; j < 2; ++j)
        acc[i][j] = wmma_bf16(af[i], bfr[j], acc[i][j]);

    wait_tiles();
    cur ^= 1;
  }

  // C/D layout: VGPR v holds row (v + 8*lh), col = lm within the 16x16 tile
  #pragma unroll
  for (int i = 0; i < 4; ++i) {
    #pragma unroll
    for (int j = 0; j < 2; ++j) {
      int gn = n0 + wn0 + j * 16 + lm;
      #pragma unroll
      for (int v = 0; v < 8; ++v) {
        int row = m0 + wm0 + i * 16 + v + 8 * lh;
        float val = acc[i][j][v];
        if (mode == 0) {
          outF[row * kH + gn] = val;
        } else {
          int b = row >> 11, s = row & 2047;
          int h = gn >> 6,  d = gn & 63;
          outS[((b * kNH + h) * kS + s) * kHD + d] = f2bf(val);
        }
      }
    }
  }
}

// ------------------------------------------------ causal flash attention bf16
// grid: (S/64, B*NH), block: 128 (4 waves). Wave w owns 16 q-rows.
// K tile staged via async-to-LDS, V tile transposed via VGPRs; double-buffered.
static constexpr int LKV = kHD + 8;  // 72 halfs = 144B rows (16B aligned)

__global__ __launch_bounds__(128) void attn_kernel(
    const u16* __restrict__ Q, const u16* __restrict__ K,
    const u16* __restrict__ V, u16* __restrict__ Oout) {
  __shared__ u16 Kt[2][64 * LKV];     // [k_row][d]   (row-major)
  __shared__ u16 Vt[2][64 * LKV];     // [d][k_row]   (transposed)
  __shared__ u16 Ps[4 * 16 * LKV];    // per-wave P staging [w][qrow][col]

  const int tid  = threadIdx.x;
  const int wave = tid >> 5;
  const int lane = tid & 31;
  const int lm   = lane & 15;
  const int lh   = lane >> 4;
  const int qt   = blockIdx.x;
  const int bh   = blockIdx.y;
  const int q0   = qt * 64 + wave * 16;

  const u16* Qb = Q + (size_t)bh * kS * kHD;
  const u16* Kb = K + (size_t)bh * kS * kHD;
  const u16* Vb = V + (size_t)bh * kS * kHD;

  // Q fragments for d-chunks [0,32) and [32,64), loaded once (A layout)
  Frag qf[2];
  const u16* qrow = Qb + (q0 + lm) * kHD;
  #pragma unroll
  for (int c = 0; c < 2; ++c) {
    qf[c].u[0] = *(const u32x4*)(qrow + c * 32 + 8 * lh);
    qf[c].u[1] = *(const u32x4*)(qrow + c * 32 + 16 + 8 * lh);
  }

  v8f o[4] = {};                  // 16x64 output accumulator (4 d-tiles)
  float mrow[8], lrow[8];
  #pragma unroll
  for (int v = 0; v < 8; ++v) { mrow[v] = -kBIG; lrow[v] = 0.f; }

  const float sl = 0.125f * 1.44269504088896f;   // (1/sqrt(HD)) * log2(e)
  const int kend = (qt + 1) * 64;

  auto fill = [&](int kb, int buf) {
    // K tile: 2 threads per key-row, 4 async 16B copies each
    int kr = tid >> 1, dp = (tid & 1) * 32;
    const u16* gk = Kb + (kb + kr) * kHD + dp;
    u16* lk = &Kt[buf][kr * LKV + dp];
    #pragma unroll
    for (int j = 0; j < 4; ++j) async_cp16(gk + 8 * j, lk + 8 * j);
    // prefetch next K block toward L2 (global_prefetch_b8)
    if (kb + 64 < kS) __builtin_prefetch(Kb + (kb + 64 + kr) * kHD + dp, 0, 0);
    // V tile transposed: 512 x 8-half segments through VGPRs
    #pragma unroll
    for (int j = 0; j < 4; ++j) {
      int seg = tid + j * 128;
      int vr = seg >> 3, d0 = (seg & 7) * 8;
      union { u32x4 u; u16 s[8]; } t;
      t.u = *(const u32x4*)(Vb + (kb + vr) * kHD + d0);
      #pragma unroll
      for (int e = 0; e < 8; ++e) Vt[buf][(d0 + e) * LKV + vr] = t.s[e];
    }
  };

  fill(0, 0);
  async_wait();
  __syncthreads();

  int cur = 0;
  for (int kb = 0; kb < kend; kb += 64) {
    if (kb + 64 < kend) fill(kb + 64, cur ^ 1);  // overlap staging with compute

    // ---- scores S' = (Q K^T) * scale * log2e, 4 tiles of 16x16
    float p[4][8];
    float cmax[8];
    #pragma unroll
    for (int v = 0; v < 8; ++v) cmax[v] = -kBIG;
    #pragma unroll
    for (int tn = 0; tn < 4; ++tn) {
      v8f sc = {};
      #pragma unroll
      for (int c = 0; c < 2; ++c) {
        Frag kf;
        const u16* kp = &Kt[cur][(tn * 16 + lm) * LKV + c * 32 + 8 * lh];
        kf.u[0] = *(const u32x4*)(kp);
        kf.u[1] = *(const u32x4*)(kp + 16);
        sc = wmma_bf16(qf[c], kf, sc);
      }
      int col = kb + tn * 16 + lm;
      #pragma unroll
      for (int v = 0; v < 8; ++v) {
        int row = q0 + v + 8 * lh;
        float s = (col <= row) ? sc[v] * sl : -kBIG;
        p[tn][v] = s;
        cmax[v]  = fmaxf(cmax[v], s);
      }
    }

    // ---- streaming softmax update per row (16 cols per lane-half)
    #pragma unroll
    for (int v = 0; v < 8; ++v) {
      float mv = cmax[v];
      #pragma unroll
      for (int off = 1; off < 16; off <<= 1)
        mv = fmaxf(mv, __shfl_xor(mv, off, 32));
      float mnew  = fmaxf(mrow[v], mv);
      float alpha = exp2f(mrow[v] - mnew);
      float rs = 0.f;
      #pragma unroll
      for (int tn = 0; tn < 4; ++tn) {
        float e = exp2f(p[tn][v] - mnew);
        p[tn][v] = e;
        rs += e;
      }
      #pragma unroll
      for (int off = 1; off < 16; off <<= 1)
        rs += __shfl_xor(rs, off, 32);
      lrow[v] = lrow[v] * alpha + rs;
      mrow[v] = mnew;
      #pragma unroll
      for (int dt = 0; dt < 4; ++dt) o[dt][v] *= alpha;
    }

    // ---- C-layout -> A-layout via per-wave LDS staging (same-wave, in-order)
    u16* pw = &Ps[wave * 16 * LKV];
    #pragma unroll
    for (int tn = 0; tn < 4; ++tn)
      #pragma unroll
      for (int v = 0; v < 8; ++v)
        pw[(v + 8 * lh) * LKV + tn * 16 + lm] = f2bf(p[tn][v]);

    Frag pf[2];
    #pragma unroll
    for (int c = 0; c < 2; ++c) {
      const u16* pp = &pw[lm * LKV + c * 32];
      pf[c].u[0] = *(const u32x4*)(pp + 8 * lh);
      pf[c].u[1] = *(const u32x4*)(pp + 16 + 8 * lh);
    }
    // ---- O += P V  (B frag from transposed V: n = d, K contiguous)
    #pragma unroll
    for (int dt = 0; dt < 4; ++dt) {
      #pragma unroll
      for (int c = 0; c < 2; ++c) {
        Frag vf;
        const u16* vp = &Vt[cur][(dt * 16 + lm) * LKV + c * 32 + 8 * lh];
        vf.u[0] = *(const u32x4*)(vp);
        vf.u[1] = *(const u32x4*)(vp + 16);
        o[dt] = wmma_bf16(pf[c], vf, o[dt]);
      }
    }

    async_wait();
    __syncthreads();
    cur ^= 1;
  }

  // ---- normalize and store to [B][S][HID] bf16
  const int b = bh >> 4, h = bh & 15;
  #pragma unroll
  for (int v = 0; v < 8; ++v) {
    float inv = 1.0f / (lrow[v] + 1e-8f);
    int row = q0 + v + 8 * lh;
    u16* orow = Oout + ((size_t)b * kS + row) * kH + h * kHD;
    #pragma unroll
    for (int dt = 0; dt < 4; ++dt)
      orow[dt * 16 + lm] = f2bf(o[dt][v] * inv);
  }
}

// ---------------------------------------------------------------------- host
extern "C" void kernel_launch(void* const* d_in, const int* in_sizes, int n_in,
                              void* d_out, int out_size, void* d_ws, size_t ws_size,
                              hipStream_t stream) {
  (void)in_sizes; (void)n_in; (void)out_size; (void)ws_size;
  const float* x  = (const float*)d_in[0];
  const float* wq = (const float*)d_in[1];
  const float* wk = (const float*)d_in[2];
  const float* wv = (const float*)d_in[3];
  const float* wo = (const float*)d_in[4];

  char* ws = (char*)d_ws;
  const size_t MB = (size_t)1 << 20;
  u16* x_bf  = (u16*)(ws + 0 * MB);   // 8 MB
  u16* wq_bf = (u16*)(ws + 8 * MB);   // 2 MB
  u16* wk_bf = (u16*)(ws + 10 * MB);
  u16* wv_bf = (u16*)(ws + 12 * MB);
  u16* wo_bf = (u16*)(ws + 14 * MB);
  u16* q_bf  = (u16*)(ws + 16 * MB);  // 8 MB each, [B][NH][S][HD]
  u16* k_bf  = (u16*)(ws + 24 * MB);
  u16* v_bf  = (u16*)(ws + 32 * MB);
  u16* a_bf  = (u16*)(ws + 40 * MB);  // attn out [B][S][HID]

  cvt_bf16_kernel<<<(kM * kH / 4) / 256, 256, 0, stream>>>(x,  x_bf,  kM * kH / 4);
  cvt_bf16_kernel<<<(kH * kH / 4) / 256, 256, 0, stream>>>(wq, wq_bf, kH * kH / 4);
  cvt_bf16_kernel<<<(kH * kH / 4) / 256, 256, 0, stream>>>(wk, wk_bf, kH * kH / 4);
  cvt_bf16_kernel<<<(kH * kH / 4) / 256, 256, 0, stream>>>(wv, wv_bf, kH * kH / 4);
  cvt_bf16_kernel<<<(kH * kH / 4) / 256, 256, 0, stream>>>(wo, wo_bf, kH * kH / 4);

  dim3 gg(kH / BN, kM / BM);          // (8, 32)
  gemm_bf16_kernel<<<gg, 256, 0, stream>>>(x_bf, wq_bf, nullptr, q_bf, 1);
  gemm_bf16_kernel<<<gg, 256, 0, stream>>>(x_bf, wk_bf, nullptr, k_bf, 1);
  gemm_bf16_kernel<<<gg, 256, 0, stream>>>(x_bf, wv_bf, nullptr, v_bf, 1);

  attn_kernel<<<dim3(kS / 64, kB * kNH), 128, 0, stream>>>(q_bf, k_bf, v_bf, a_bf);

  gemm_bf16_kernel<<<gg, 256, 0, stream>>>(a_bf, wo_bf, (float*)d_out, nullptr, 0);
}
